// AttentionKVSplitted_51135880626361
// MI455X (gfx1250) — matrix operand order repaired and checked by reading, not compiled, not run on hardware
//
#include <hip/hip_runtime.h>
#include <hip/hip_bf16.h>

// Problem constants (fixed by reference setup)
#define BB 8
#define NN 2048
#define MM 32768
#define QDIM 512
#define DRR 256
#define DDD 512
#define HH 8
#define DHH 64
#define INNERD 512
#define TOPK 8

typedef __attribute__((ext_vector_type(16))) __bf16 v16bf;
typedef __attribute__((ext_vector_type(8)))  float  v8f;
typedef __attribute__((ext_vector_type(4)))  unsigned int u32x4;

union FragU { v16bf v; u32x4 q[2]; };

__device__ __forceinline__ unsigned short f2bf(float f) {
  unsigned int x = __float_as_uint(f);
  x += 0x7FFFu + ((x >> 16) & 1u);   // round-to-nearest-even
  return (unsigned short)(x >> 16);
}
__device__ __forceinline__ unsigned int f2bf2(float lo, float hi) {
  return (unsigned int)f2bf(lo) | ((unsigned int)f2bf(hi) << 16);
}

// ---------------------------------------------------------------------------
// Bulk fp32 -> bf16 conversion (8 elements / thread)
// ---------------------------------------------------------------------------
__global__ __launch_bounds__(256) void cvt_bf16_kernel(
    const float* __restrict__ in, unsigned short* __restrict__ outp)
{
  size_t i = ((size_t)blockIdx.x * 256 + threadIdx.x) * 8;
  float4 a = *(const float4*)(in + i);
  float4 b = *(const float4*)(in + i + 4);
  u32x4 o;
  o.x = f2bf2(a.x, a.y); o.y = f2bf2(a.z, a.w);
  o.z = f2bf2(b.x, b.y); o.w = f2bf2(b.z, b.w);
  *(u32x4*)(outp + i) = o;
}

// ---------------------------------------------------------------------------
// 512x512 fp32 -> transposed bf16 (Wt[n*512+k] = W[k*512+n]); 32x32 LDS tiles
// ---------------------------------------------------------------------------
__global__ __launch_bounds__(256) void transpose_cvt_kernel(
    const float* __restrict__ W, unsigned short* __restrict__ Wt)
{
  __shared__ float tile[32][33];
  int bx = blockIdx.x & 15, by = blockIdx.x >> 4;   // n-tile, k-tile
  int t = threadIdx.x;
  int r  = t >> 3;            // 0..31
  int c0 = (t & 7) << 2;      // 0,4,...,28
  float4 v = *(const float4*)(W + (size_t)(by * 32 + r) * 512 + bx * 32 + c0);
  tile[r][c0] = v.x; tile[r][c0 + 1] = v.y; tile[r][c0 + 2] = v.z; tile[r][c0 + 3] = v.w;
  __syncthreads();
  // output row n = bx*32 + r, cols k = by*32 + c0..c0+3
  ushort4 o;
  o.x = f2bf(tile[c0 + 0][r]);
  o.y = f2bf(tile[c0 + 1][r]);
  o.z = f2bf(tile[c0 + 2][r]);
  o.w = f2bf(tile[c0 + 3][r]);
  *(ushort4*)(Wt + (size_t)(bx * 32 + r) * 512 + by * 32 + c0) = o;
}

// ---------------------------------------------------------------------------
// GEMM: C[16384,512] = A[16384,512](bf16) @ B(bf16, pre-transposed N x K) +bias
// Block tile 128x64, 256 threads = 8 waves (4x2), wave = 32x32 = 2x2 WMMA.
// LDS tiles in bf16, 80-byte row stride (16B aligned, conflict-free).
// Fragments load as 2x ds_load_b128 each (layouts are contiguous runs).
// ---------------------------------------------------------------------------
template<bool BIAS>
__global__ __launch_bounds__(256) void gemm_bf16_wmma(
    const unsigned short* __restrict__ A,   // M x 512, bf16 row-major
    const unsigned short* __restrict__ Bt,  // 512(N) x 512(K), bf16
    const float* __restrict__ bias, float* __restrict__ C)
{
  __shared__ unsigned short Asm[128 * 40];  // 128 rows x 32 k (+pad) = 10240 B
  __shared__ unsigned short Bsm[64 * 40];   //  64 n    x 32 k (+pad) =  5120 B

  const int t     = threadIdx.x;
  const int lane  = t & 31;
  const int wave  = t >> 5;
  const int half  = lane >> 4;
  const int lm    = lane & 15;
  const int waveM = wave & 3;
  const int waveN = wave >> 2;
  const long long bm = (long long)blockIdx.x * 128;
  const int bn = blockIdx.y * 64;

  // staging indices (constant across k-loop)
  const int arow0 = t >> 2;               // 0..63
  const int ac    = (t & 3) << 3;         // ushort offset 0,8,16,24
  const int brow  = t >> 2;               // n within tile 0..63
  const unsigned short* Abase = A + (bm + arow0) * 512 + ac;
  const unsigned short* Bbase = Bt + (long long)(bn + brow) * 512 + ac;

  v8f zero = {0.f,0.f,0.f,0.f,0.f,0.f,0.f,0.f};
  v8f acc[2][2];
  #pragma unroll
  for (int i = 0; i < 2; ++i)
    #pragma unroll
    for (int j = 0; j < 2; ++j) acc[i][j] = zero;

  for (int k0 = 0; k0 < 512; k0 += 32) {
    __syncthreads();
    // Stage A tile: 128x32 bf16 = 512 x b128 -> 2 per thread
    #pragma unroll
    for (int rr = 0; rr < 2; ++rr) {
      u32x4 v = *(const u32x4*)(Abase + (size_t)rr * 64 * 512 + k0);
      *(u32x4*)&Asm[(arow0 + 64 * rr) * 40 + ac] = v;
    }
    // Stage B tile: 64x32 bf16 = 256 x b128 -> 1 per thread
    {
      u32x4 v = *(const u32x4*)(Bbase + k0);
      *(u32x4*)&Bsm[brow * 40 + ac] = v;
    }
    __syncthreads();

    FragU aF[2], bF[2];
    #pragma unroll
    for (int i = 0; i < 2; ++i) {
      const unsigned short* ab = &Asm[(waveM * 32 + i * 16 + lm) * 40 + 8 * half];
      aF[i].q[0] = *(const u32x4*)(ab);        // K = 8*half .. +7
      aF[i].q[1] = *(const u32x4*)(ab + 16);   // K = 16+8*half .. +7
    }
    #pragma unroll
    for (int j = 0; j < 2; ++j) {
      const unsigned short* bb = &Bsm[(waveN * 32 + j * 16 + lm) * 40 + 8 * half];
      bF[j].q[0] = *(const u32x4*)(bb);        // K = 8*half .. +7
      bF[j].q[1] = *(const u32x4*)(bb + 16);   // K = 16+8*half .. +7
    }
    #pragma unroll
    for (int i = 0; i < 2; ++i)
      #pragma unroll
      for (int j = 0; j < 2; ++j)
        acc[i][j] = __builtin_amdgcn_wmma_f32_16x16x32_bf16(
            false, aF[i].v, false, bF[j].v, (short)0, acc[i][j], false, false);
  }

  // C/D layout: VGPR r -> M = r + 8*half, lane&15 -> N
  #pragma unroll
  for (int i = 0; i < 2; ++i) {
    #pragma unroll
    for (int j = 0; j < 2; ++j) {
      int col = bn + waveN * 32 + j * 16 + lm;
      float bval = BIAS ? bias[col] : 0.0f;
      #pragma unroll
      for (int r = 0; r < 8; ++r) {
        long long row = bm + waveM * 32 + i * 16 + r + 8 * half;
        C[row * 512 + col] = acc[i][j][r] + bval;
      }
    }
  }
}

// ---------------------------------------------------------------------------
// e0[b,:] = (x[b,0,:] @ Wq) @ We   (exact fp32; feeds ranking path)
// ---------------------------------------------------------------------------
__global__ __launch_bounds__(256) void e0_kernel(
    const float* __restrict__ x, const float* __restrict__ Wq,
    const float* __restrict__ We, float* __restrict__ e0)
{
  __shared__ float xs[512];
  __shared__ float ts[512];
  int b = blockIdx.x, t = threadIdx.x;
  const float* xr = x + (long long)b * NN * QDIM;   // row n=0
  xs[t] = xr[t]; xs[t + 256] = xr[t + 256];
  __syncthreads();
  #pragma unroll
  for (int oo = 0; oo < 2; ++oo) {
    int o = t + 256 * oo;
    float s = 0.f;
    for (int i = 0; i < 512; ++i) s = fmaf(xs[i], Wq[(long long)i * 512 + o], s);
    ts[o] = s;
  }
  __syncthreads();
  float s = 0.f;
  for (int i = 0; i < 512; ++i) s = fmaf(ts[i], We[i * 256 + t], s);
  e0[b * 256 + t] = s;
}

// ---------------------------------------------------------------------------
// dist2[b,m] = || context[b,m,:256] - e0[b,:] ||^2   (wave per row)
// ---------------------------------------------------------------------------
__global__ __launch_bounds__(256) void dist_kernel(
    const float* __restrict__ ctx, const float* __restrict__ e0,
    float* __restrict__ dist)
{
  int t = threadIdx.x, lane = t & 31, wave = t >> 5;
  long long r = (long long)blockIdx.x * 8 + wave;   // r = b*M + m
  int b = (int)(r >> 15);
  const float4* rep = (const float4*)(ctx + r * DDD);
  const float4* e4  = (const float4*)(e0 + b * 256);
  float s = 0.f;
  #pragma unroll
  for (int p = 0; p < 2; ++p) {
    float4 a = rep[lane + 32 * p];
    float4 c = e4[lane + 32 * p];
    float dx = a.x - c.x, dy = a.y - c.y, dz = a.z - c.z, dw = a.w - c.w;
    s += dx * dx + dy * dy + dz * dz + dw * dw;
  }
  #pragma unroll
  for (int off = 16; off; off >>= 1) s += __shfl_xor(s, off, 32);
  if (lane == 0) dist[r] = s;
}

// ---------------------------------------------------------------------------
// top-8 smallest (dist, idx) per batch; tie-break lower index (matches top_k)
// ---------------------------------------------------------------------------
__global__ __launch_bounds__(256) void topk_kernel(
    const float* __restrict__ dist, int* __restrict__ idxout)
{
  __shared__ float cd[256];
  __shared__ int   ci[256];
  __shared__ int   ct[256];
  int b = blockIdx.x, t = threadIdx.x;
  const float* d = dist + (long long)b * MM;

  float bd[8]; int bi[8];
  #pragma unroll
  for (int q = 0; q < 8; ++q) { bd[q] = 3.4e38f; bi[q] = 0x7fffffff; }

  for (int s2 = 0; s2 < MM / 256; ++s2) {
    int i = t + 256 * s2;
    float dv = d[i];
    if (dv < bd[7] || (dv == bd[7] && i < bi[7])) {
      int p = 7;
      while (p > 0 && (dv < bd[p - 1] || (dv == bd[p - 1] && i < bi[p - 1]))) {
        bd[p] = bd[p - 1]; bi[p] = bi[p - 1]; --p;
      }
      bd[p] = dv; bi[p] = i;
    }
  }

  int ptr = 0;
  for (int round = 0; round < TOPK; ++round) {
    cd[t] = (ptr < 8) ? bd[ptr] : 3.4e38f;
    ci[t] = (ptr < 8) ? bi[ptr] : 0x7fffffff;
    ct[t] = t;
    __syncthreads();
    for (int g = 128; g; g >>= 1) {
      if (t < g) {
        float d2 = cd[t + g]; int i2 = ci[t + g];
        if (d2 < cd[t] || (d2 == cd[t] && i2 < ci[t])) {
          cd[t] = d2; ci[t] = i2; ct[t] = ct[t + g];
        }
      }
      __syncthreads();
    }
    if (t == 0) idxout[b * TOPK + round] = ci[0];
    if (t == ct[0]) ++ptr;
    __syncthreads();
  }
}

// ---------------------------------------------------------------------------
// gather + k = labels@Wk, v = reps@Wv   (block per (b,j), fp32)
// ---------------------------------------------------------------------------
__global__ __launch_bounds__(256) void kv_kernel(
    const float* __restrict__ ctx, const int* __restrict__ idx,
    const float* __restrict__ Wk, const float* __restrict__ Wv,
    float* __restrict__ kout, float* __restrict__ vout)
{
  __shared__ float row[512];
  int bj = blockIdx.x, t = threadIdx.x;
  int b = bj >> 3;
  int id = idx[bj];
  const float* src = ctx + ((long long)b * MM + id) * DDD;
  row[t] = src[t]; row[t + 256] = src[t + 256];
  __syncthreads();
  #pragma unroll
  for (int oo = 0; oo < 2; ++oo) {
    int o = t + 256 * oo;
    float sk = 0.f, sv = 0.f;
    for (int i = 0; i < 256; ++i) {
      sk = fmaf(row[256 + i], Wk[(long long)i * 512 + o], sk);  // labels
      sv = fmaf(row[i],       Wv[(long long)i * 512 + o], sv);  // reps
    }
    kout[(long long)bj * 512 + o] = sk;
    vout[(long long)bj * 512 + o] = sv;
  }
}

// ---------------------------------------------------------------------------
// attention: wave per (b,n) row; k/v shared across n; output bf16 for GEMM #2
// ---------------------------------------------------------------------------
__global__ __launch_bounds__(256) void attn_kernel(
    const float* __restrict__ q, const float* __restrict__ kb,
    const float* __restrict__ vb, unsigned short* __restrict__ outp)
{
  __shared__ float qs[8][512];
  __shared__ float sm[8][64];
  int t = threadIdx.x, lane = t & 31, wave = t >> 5;
  long long r = (long long)blockIdx.x * 8 + wave;   // 0..16383
  int b = (int)(r >> 11);                            // N = 2048
  const float* qr = q + r * 512;
  #pragma unroll
  for (int p = 0; p < 16; ++p) qs[wave][lane + 32 * p] = qr[lane + 32 * p];
  __syncthreads();

  #pragma unroll
  for (int pp = 0; pp < 2; ++pp) {
    int p = lane + 32 * pp;            // 0..63, p = h*8 + j
    int h = p >> 3, j = p & 7;
    const float* kr = kb + ((long long)b * TOPK + j) * 512 + h * 64;
    const float* qh = &qs[wave][h * 64];
    float s = 0.f;
    #pragma unroll 8
    for (int d2 = 0; d2 < 64; ++d2) s = fmaf(qh[d2], kr[d2], s);
    sm[wave][p] = s * 0.125f;          // DH^-0.5
  }
  __syncthreads();

  if (lane < 8) {                      // lane == head
    float* sh = &sm[wave][lane * 8];
    float mx = sh[0];
    #pragma unroll
    for (int j = 1; j < 8; ++j) mx = fmaxf(mx, sh[j]);
    float e[8], sum = 0.f;
    #pragma unroll
    for (int j = 0; j < 8; ++j) { e[j] = __expf(sh[j] - mx); sum += e[j]; }
    float inv = 1.0f / sum;
    #pragma unroll
    for (int j = 0; j < 8; ++j) sh[j] = e[j] * inv;
  }
  __syncthreads();

  #pragma unroll
  for (int p = 0; p < 16; ++p) {
    int o = lane + 32 * p;             // 0..511
    int h = o >> 6;
    float s = 0.f;
    #pragma unroll
    for (int j = 0; j < 8; ++j)
      s = fmaf(sm[wave][h * 8 + j], vb[((long long)b * TOPK + j) * 512 + o], s);
    outp[r * 512 + o] = f2bf(s);
  }
}

// ---------------------------------------------------------------------------
extern "C" void kernel_launch(void* const* d_in, const int* in_sizes, int n_in,
                              void* d_out, int out_size, void* d_ws, size_t ws_size,
                              hipStream_t stream) {
  const float* x    = (const float*)d_in[0];
  const float* ctx  = (const float*)d_in[1];
  const float* Wq   = (const float*)d_in[2];
  const float* Wk   = (const float*)d_in[3];
  const float* Wv   = (const float*)d_in[4];
  const float* We   = (const float*)d_in[5];
  const float* Wout = (const float*)d_in[6];
  const float* bout = (const float*)d_in[7];
  float* out = (float*)d_out;

  char* ws = (char*)d_ws;
  size_t off = 0;
  float*          qbuf   = (float*)(ws + off);          off += (size_t)BB * NN * INNERD * 4;  // 33.5 MB
  unsigned short* xbf    = (unsigned short*)(ws + off); off += (size_t)BB * NN * QDIM * 2;    // 16.7 MB
  unsigned short* abf    = (unsigned short*)(ws + off); off += (size_t)BB * NN * INNERD * 2;  // 16.7 MB
  unsigned short* WqT    = (unsigned short*)(ws + off); off += (size_t)512 * 512 * 2;
  unsigned short* WoutT  = (unsigned short*)(ws + off); off += (size_t)512 * 512 * 2;
  float*          e0buf  = (float*)(ws + off);          off += 8192;
  float*          dbuf   = (float*)(ws + off);          off += (size_t)BB * MM * 4;           // 1 MB
  int*            idxbuf = (int*)(ws + off);            off += 4096;
  float*          kbuf   = (float*)(ws + off);          off += (size_t)BB * TOPK * INNERD * 4;
  float*          vbuf   = (float*)(ws + off);          off += (size_t)BB * TOPK * INNERD * 4;

  // 0) one-time precision/layout prep
  cvt_bf16_kernel<<<(BB * NN * QDIM) / (256 * 8), 256, 0, stream>>>(x, xbf);
  transpose_cvt_kernel<<<256, 256, 0, stream>>>(Wq, WqT);
  transpose_cvt_kernel<<<256, 256, 0, stream>>>(Wout, WoutT);

  // 1) q = x @ Wq  (bf16 WMMA, fp32 accum)
  gemm_bf16_wmma<false><<<dim3(128, 8), 256, 0, stream>>>(xbf, WqT, nullptr, qbuf);
  // 2) e0 (exact fp32)
  e0_kernel<<<BB, 256, 0, stream>>>(x, Wq, We, e0buf);
  // 3) squared distances
  dist_kernel<<<(BB * MM) / 8, 256, 0, stream>>>(ctx, e0buf, dbuf);
  // 4) top-8 indices per batch
  topk_kernel<<<BB, 256, 0, stream>>>(dbuf, idxbuf);
  // 5) gather + k/v projections
  kv_kernel<<<BB * TOPK, 256, 0, stream>>>(ctx, idxbuf, Wk, Wv, kbuf, vbuf);
  // 6) attention -> bf16
  attn_kernel<<<(BB * NN) / 8, 256, 0, stream>>>(qbuf, kbuf, vbuf, abf);
  // 7) final projection + bias (bf16 WMMA)
  gemm_bf16_wmma<true><<<dim3(128, 8), 256, 0, stream>>>(abf, WoutT, bout, out);
}
